// RGCN_90374701843213
// MI455X (gfx1250) — compile-verified, matrix operand order checked by reading, not run on hardware
//
#include <hip/hip_runtime.h>
#include <hip/hip_bf16.h>

// ---------------------------------------------------------------------------
// RGCN forward for MI455X (gfx1250, wave32, WMMA, async global->LDS copies).
// Dominant work: C[8192,1152] = x[8192,8192] @ W1cat[8192,1152] with
// v_wmma_f32_16x16x32_bf16; tiles staged with GLOBAL_LOAD_ASYNC_TO_LDS_B128
// (ASYNCcnt, double buffered) when the toolchain exposes the builtin.
// ---------------------------------------------------------------------------

typedef __attribute__((ext_vector_type(16))) __bf16        v16bf;
typedef __attribute__((ext_vector_type(8)))  float         v8f;
typedef __attribute__((ext_vector_type(4)))  unsigned int  v4u;

union FragBF {
    v16bf f;
    v4u   q[2];   // two 16-byte halves for LDS b128 loads
};

__device__ __forceinline__ unsigned short f2bf(float x) {
    unsigned u = __float_as_uint(x);
    unsigned r = u + 0x7FFFu + ((u >> 16) & 1u);   // round-to-nearest-even
    return (unsigned short)(r >> 16);
}
__device__ __forceinline__ unsigned pack2(float a, float b) {
    return (unsigned)f2bf(a) | ((unsigned)f2bf(b) << 16);
}

// ---- async global->LDS (CDNA5): 16 B per lane, tracked by ASYNCcnt --------
#if defined(__has_builtin)
#if __has_builtin(__builtin_amdgcn_global_load_async_to_lds_b128)
#define HAVE_ASYNC_LDS 1
#endif
#endif

// Builtin parameter type (from hipcc diagnostic): AS-qualified pointer to
// 'int __vector(4)'.
typedef int b128vec __attribute__((vector_size(16)));
typedef __attribute__((address_space(1))) b128vec* gptr128_t;
typedef __attribute__((address_space(3))) b128vec* lptr128_t;

__device__ __forceinline__ void async_copy_b128(const unsigned short* g,
                                                unsigned short* l) {
#ifdef HAVE_ASYNC_LDS
    // Global aperture is identity; LDS offset = low 32 bits of the flat addr.
    __builtin_amdgcn_global_load_async_to_lds_b128(
        (gptr128_t)(unsigned long long)g,
        (lptr128_t)(unsigned)(unsigned long long)l,
        0, 0);
#else
    *(v4u*)l = *(const v4u*)g;     // fallback: global_load_b128 + ds_store_b128
#endif
}

__device__ __forceinline__ void wait_async_prev_tile() {
#ifdef HAVE_ASYNC_LDS
#if __has_builtin(__builtin_amdgcn_s_wait_asynccnt)
    __builtin_amdgcn_s_wait_asynccnt(4);   // 4 newer ops (next tile) may remain
#else
    asm volatile("s_wait_asynccnt 0x4" ::: "memory");
#endif
#endif
}
__device__ __forceinline__ void wait_async_all() {
#ifdef HAVE_ASYNC_LDS
#if __has_builtin(__builtin_amdgcn_s_wait_asynccnt)
    __builtin_amdgcn_s_wait_asynccnt(0);
#else
    asm volatile("s_wait_asynccnt 0x0" ::: "memory");
#endif
#endif
}

// ---------------------------------------------------------------------------
// x (f32) -> bf16, 4 elements per thread.
// ---------------------------------------------------------------------------
__global__ __launch_bounds__(256)
void cvt_f32_to_bf16(const float* __restrict__ in,
                     unsigned short* __restrict__ outp, long long n4)
{
    long long gid = (long long)blockIdx.x * blockDim.x + threadIdx.x;
    if (gid >= n4) return;
    float4 v = ((const float4*)in)[gid];
    uint2 o;
    o.x = pack2(v.x, v.y);
    o.y = pack2(v.z, v.w);
    ((uint2*)outp)[gid] = o;
}

// ---------------------------------------------------------------------------
// W_r = sum_b comp[r,b] * basis[b]  plus root appended as the last Dout cols.
// Wcat layout: [K rows][R*Dout + Dout cols] f32, row-major.
// ---------------------------------------------------------------------------
__global__ __launch_bounds__(256)
void build_wcat(const float* __restrict__ basis,   // [B, K, Dout]
                const float* __restrict__ comp,    // [R, B]
                const float* __restrict__ root,    // [K, Dout]
                float* __restrict__ Wcat,          // [K, R*Dout + Dout]
                int K, int Dout, int R, int B)
{
    __shared__ float sc[256];                      // R*B <= 240
    for (int t = threadIdx.x; t < R * B; t += blockDim.x) sc[t] = comp[t];
    __syncthreads();

    long long gid = (long long)blockIdx.x * blockDim.x + threadIdx.x;
    if (gid >= (long long)K * Dout) return;
    int k = (int)(gid / Dout);
    int o = (int)(gid % Dout);

    float acc[8];
    #pragma unroll
    for (int r = 0; r < 8; ++r) acc[r] = 0.0f;

    for (int b = 0; b < B; ++b) {
        float bv = basis[((long long)b * K + k) * Dout + o];
        #pragma unroll
        for (int r = 0; r < 8; ++r) acc[r] += sc[r * B + b] * bv;
    }
    int ld = R * Dout + Dout;
    #pragma unroll
    for (int r = 0; r < 8; ++r)
        Wcat[(long long)k * ld + r * Dout + o] = acc[r];
    Wcat[(long long)k * ld + R * Dout + o] = root[(long long)k * Dout + o];
}

// ---------------------------------------------------------------------------
// Transpose + convert: in[K][Ncols] f32 -> outT[NcolsPad][K] bf16.
// Pad columns (c >= Ncols) are zero-filled. 32x32 LDS tile.
// ---------------------------------------------------------------------------
__global__ __launch_bounds__(256)
void transpose_cvt_bf16(const float* __restrict__ in,
                        unsigned short* __restrict__ outT,
                        int K, int Ncols, int NcolsPad)
{
    __shared__ float tile[32][33];
    int kb = blockIdx.x * 32, cb = blockIdx.y * 32;
    int tx = threadIdx.x & 31, ty = threadIdx.x >> 5;
    #pragma unroll
    for (int i = 0; i < 4; ++i) {
        int k = kb + ty + i * 8, c = cb + tx;
        tile[ty + i * 8][tx] =
            (k < K && c < Ncols) ? in[(long long)k * Ncols + c] : 0.0f;
    }
    __syncthreads();
    #pragma unroll
    for (int i = 0; i < 4; ++i) {
        int c = cb + ty + i * 8, k = kb + tx;
        if (c < NcolsPad && k < K)
            outT[(long long)c * K + k] = f2bf(tile[tx][ty + i * 8]);
    }
}

// ---------------------------------------------------------------------------
// bf16 WMMA GEMM: C[M,Ncols] = A[M,K] @ BT[col][k]  (+ bias[col]).
// A: bf16 row-major [M][K]; BT: bf16 pre-transposed [>=NcolsUsed][K].
// Block tile 128x128, BK=32, double-buffered LDS filled by async copies.
// 256 threads = 8 waves; wave grid 4(M) x 2(N); each wave: 2x4 WMMA tiles.
// LDS row stride 40 halves (80 B) keeps fragment b128 loads 16B-aligned and
// bank-spread; fragments follow the 16-bit A-matrix lane layout
// (lanes 0-15: K 0-7/16-23, lanes 16-31: K 8-15/24-31).
// Requires M % 128 == 0, K % 32 == 0.
// ---------------------------------------------------------------------------
#define BM 128
#define BN 128
#define BKK 32
#define LDSROW 40

__device__ __forceinline__ void issue_tile(
    const unsigned short* __restrict__ A, const unsigned short* __restrict__ BT,
    unsigned short* As, unsigned short* Bs,
    int blockM, int blockN, int lda, int K, int kt, int wave, int lane)
{
    #pragma unroll
    for (int i = 0; i < 2; ++i) {
        int seg  = wave * 64 + i * 32 + lane;   // 0..511 across the block
        int row  = seg >> 2;                    // 0..127
        int part = (seg & 3) << 3;              // 0,8,16,24 halves (16 B each)
        async_copy_b128(A  + (long long)(blockM + row) * lda + kt + part,
                        &As[row * LDSROW + part]);
        async_copy_b128(BT + (long long)(blockN + row) * K   + kt + part,
                        &Bs[row * LDSROW + part]);
    }
}

__global__ __launch_bounds__(256)
void gemm_bf16_wmma(const unsigned short* __restrict__ A,
                    const unsigned short* __restrict__ BT,
                    const float* __restrict__ bias,   // may be nullptr
                    float* __restrict__ C,
                    int M, int Ncols, int K, int lda, int ldc)
{
    __shared__ __align__(16) unsigned short As[2][BM * LDSROW];
    __shared__ __align__(16) unsigned short Bs[2][BN * LDSROW];

    const int tid    = threadIdx.x;
    const int lane   = tid & 31;
    const int wave   = tid >> 5;
    const int wm     = wave & 3;              // 32-row slab
    const int wn     = wave >> 2;             // 64-col slab
    const int blockM = blockIdx.y * BM;
    const int blockN = blockIdx.x * BN;
    const int laneLo = lane & 15;
    const int kSel   = (lane < 16) ? 0 : 8;

    v8f acc[2][4];
    #pragma unroll
    for (int i = 0; i < 2; ++i)
        #pragma unroll
        for (int j = 0; j < 4; ++j)
            #pragma unroll
            for (int e = 0; e < 8; ++e) acc[i][j][e] = 0.0f;

    const int nT = K / BKK;
    issue_tile(A, BT, As[0], Bs[0], blockM, blockN, lda, K, 0, wave, lane);

    for (int t = 0; t < nT; ++t) {
        const int cur = t & 1;
        if (t + 1 < nT) {
            issue_tile(A, BT, As[cur ^ 1], Bs[cur ^ 1],
                       blockM, blockN, lda, K, (t + 1) * BKK, wave, lane);
            wait_async_prev_tile();
        } else {
            wait_async_all();
        }
        if (t + 2 < nT)   // keep L2 warm two tiles ahead
            __builtin_prefetch(A + (long long)(blockM + (lane << 2)) * lda
                                 + (t + 2) * BKK, 0, 1);
        __syncthreads();

        FragBF af[2];
        #pragma unroll
        for (int mt = 0; mt < 2; ++mt) {
            const unsigned short* p =
                &As[cur][(wm * 32 + mt * 16 + laneLo) * LDSROW];
            af[mt].q[0] = *(const v4u*)(p + kSel);       // K 0-7  / 8-15
            af[mt].q[1] = *(const v4u*)(p + 16 + kSel);  // K 16-23 / 24-31
        }
        #pragma unroll
        for (int nt = 0; nt < 4; ++nt) {
            FragBF bf_;
            const unsigned short* p =
                &Bs[cur][(wn * 64 + nt * 16 + laneLo) * LDSROW];
            bf_.q[0] = *(const v4u*)(p + kSel);
            bf_.q[1] = *(const v4u*)(p + 16 + kSel);
            #pragma unroll
            for (int mt = 0; mt < 2; ++mt) {
                acc[mt][nt] = __builtin_amdgcn_wmma_f32_16x16x32_bf16(
                    false, af[mt].f, false, bf_.f,
                    (short)0, acc[mt][nt], false, false);
            }
        }
        __syncthreads();   // all waves done reading before buffer reuse
    }

    // ---- epilogue: C/D lane layout (lanes 0-15: M+0..7, 16-31: M+8..15) --
    #pragma unroll
    for (int mt = 0; mt < 2; ++mt) {
        #pragma unroll
        for (int nt = 0; nt < 4; ++nt) {
            int col = blockN + wn * 64 + nt * 16 + laneLo;
            if (col < Ncols) {
                float badd  = bias ? bias[col] : 0.0f;
                int   rbase = blockM + wm * 32 + mt * 16 + ((lane < 16) ? 0 : 8);
                #pragma unroll
                for (int j = 0; j < 8; ++j)
                    C[(long long)(rbase + j) * ldc + col] = acc[mt][nt][j] + badd;
            }
        }
    }
}

// ---------------------------------------------------------------------------
// Edge scatter: msg = xW[src, r*D + o]; atomic segment sums per (dst, rel).
// ---------------------------------------------------------------------------
__global__ __launch_bounds__(256)
void edge_agg(const int* __restrict__ ei,    // [2, E]
              const int* __restrict__ et,    // [E]
              const float* __restrict__ xW, int ldxw,
              float* __restrict__ acc, float* __restrict__ cnt,
              int D, int E, int R)
{
    long long gid = (long long)blockIdx.x * blockDim.x + threadIdx.x;
    if (gid >= (long long)E * D) return;
    int e = (int)(gid / D);
    int o = (int)(gid % D);
    int s = ei[e];
    int d = ei[E + e];
    int r = et[e];
    float v = xW[(long long)s * ldxw + r * D + o];
    atomicAdd(&acc[((long long)d * R + r) * D + o], v);
    if (o == 0) atomicAdd(&cnt[d * R + r], 1.0f);
}

// ---------------------------------------------------------------------------
// out_bf16[i,o] = act( sum_r acc/max(cnt,1) + xW[i,rootCol+o] + bias[o] )
// ---------------------------------------------------------------------------
__global__ __launch_bounds__(256)
void finalize_agg(const float* __restrict__ acc, const float* __restrict__ cnt,
                  const float* __restrict__ xW, int ldxw, int rootCol,
                  const float* __restrict__ bias,
                  unsigned short* __restrict__ out,   // bf16
                  int N, int D, int R, int doRelu)
{
    long long gid = (long long)blockIdx.x * blockDim.x + threadIdx.x;
    if (gid >= (long long)N * D) return;
    int i = (int)(gid / D);
    int o = (int)(gid % D);
    float s = 0.0f;
    #pragma unroll
    for (int r = 0; r < 8; ++r) {
        float c = cnt[i * R + r];
        s += acc[((long long)i * R + r) * D + o] / fmaxf(c, 1.0f);
    }
    s += xW[(long long)i * ldxw + rootCol + o] + bias[o];
    out[(long long)i * D + o] = f2bf(doRelu ? fmaxf(s, 0.0f) : s);
}

__global__ __launch_bounds__(256)
void zero_f32(float* __restrict__ p, long long n)
{
    long long gid = (long long)blockIdx.x * blockDim.x + threadIdx.x;
    if (gid < n) p[gid] = 0.0f;
}

// ---------------------------------------------------------------------------
extern "C" void kernel_launch(void* const* d_in, const int* in_sizes, int n_in,
                              void* d_out, int out_size, void* d_ws, size_t ws_size,
                              hipStream_t stream)
{
    const float* x      = (const float*)d_in[0];
    const int*   ei     = (const int*)  d_in[1];
    const int*   et     = (const int*)  d_in[2];
    const float* basis1 = (const float*)d_in[3];
    const float* comp1  = (const float*)d_in[4];
    const float* root1  = (const float*)d_in[5];
    const float* bias1  = (const float*)d_in[6];
    const float* basis2 = (const float*)d_in[7];
    const float* comp2  = (const float*)d_in[8];
    const float* root2  = (const float*)d_in[9];
    const float* bias2  = (const float*)d_in[10];
    const float* clfw   = (const float*)d_in[11];
    const float* clfb   = (const float*)d_in[12];
    float* out = (float*)d_out;
    char*  base = (char*)d_ws;

    const int N = 8192, E = 262144, R = 8, B = 30, H = 128, O = 64;

    // ---- workspace (byte offsets, all 16B aligned); conv2 aliases x_bf ----
    unsigned short* x_bf  = (unsigned short*)(base);                 // 134,217,728 B
    unsigned short* BT1   = (unsigned short*)(base + 134217728LL);   //  18,874,368 B [1152][8192]
    float*          xW1   = (float*)         (base + 153092096LL);   //  37,748,736 B [8192][1152]
    float*          W1cat = xW1;              // alias: dead before gemm1 writes xW1
    float*          acc1  = (float*)         (base + 190840832LL);   //  33,554,432 B
    float*          cnt1  = (float*)         (base + 224395264LL);   //     262,144 B
    unsigned short* h_bf  = (unsigned short*)(base + 224657408LL);   //   2,097,152 B
    // conv2 region (x_bf dead after gemm1):
    float*          W2cat = (float*)         (base);                 //     294,912 B [128][576]
    unsigned short* BT2   = (unsigned short*)(base + 294912LL);      //     163,840 B [640][128]
    float*          xW2   = (float*)         (base + 458752LL);      //  18,874,368 B [8192][576]
    float*          acc2  = (float*)         (base + 19333120LL);    //  16,777,216 B
    float*          cnt2  = (float*)         (base + 36110336LL);    //     262,144 B
    unsigned short* h2_bf = (unsigned short*)(base + 36372480LL);    //   1,048,576 B
    unsigned short* clfwT = (unsigned short*)(base + 37421056LL);    //      16,384 B [128][64]

    // ======================= conv1 =======================
    cvt_f32_to_bf16<<<16777216 / 256, 256, 0, stream>>>(x, x_bf, 16777216LL);
    build_wcat<<<(N * H) / 256, 256, 0, stream>>>(basis1, comp1, root1, W1cat,
                                                  N, H, R, B);
    {
        dim3 g(N / 32, 1152 / 32);
        transpose_cvt_bf16<<<g, 256, 0, stream>>>(W1cat, BT1, N, 1152, 1152);
    }
    {
        dim3 g(1152 / BN, N / BM);   // 9 x 64
        gemm_bf16_wmma<<<g, 256, 0, stream>>>(x_bf, BT1, nullptr, xW1,
                                              N, 1152, N, N, 1152);
    }
    zero_f32<<<(8388608 + 65536) / 256, 256, 0, stream>>>(acc1, 8388608LL + 65536LL);
    edge_agg<<<((long long)E * H) / 256, 256, 0, stream>>>(ei, et, xW1, 1152,
                                                           acc1, cnt1, H, E, R);
    finalize_agg<<<(N * H) / 256, 256, 0, stream>>>(acc1, cnt1, xW1, 1152, 1024,
                                                    bias1, h_bf, N, H, R, 1);

    // ======================= conv2 =======================
    build_wcat<<<(H * O) / 256, 256, 0, stream>>>(basis2, comp2, root2, W2cat,
                                                  H, O, R, B);
    {
        dim3 g(H / 32, 640 / 32);
        transpose_cvt_bf16<<<g, 256, 0, stream>>>(W2cat, BT2, H, 576, 640);
    }
    {
        dim3 g((576 + BN - 1) / BN, N / BM);   // 5 x 64
        gemm_bf16_wmma<<<g, 256, 0, stream>>>(h_bf, BT2, nullptr, xW2,
                                              N, 576, H, H, 576);
    }
    zero_f32<<<(4194304 + 65536) / 256, 256, 0, stream>>>(acc2, 4194304LL + 65536LL);
    edge_agg<<<((long long)E * O) / 256, 256, 0, stream>>>(ei, et, xW2, 576,
                                                           acc2, cnt2, O, E, R);
    finalize_agg<<<(N * O) / 256, 256, 0, stream>>>(acc2, cnt2, xW2, 576, 512,
                                                    bias2, h2_bf, N, O, R, 0);

    // ======================= classifier =======================
    {
        dim3 g(O / 32, 128 / 32);
        transpose_cvt_bf16<<<g, 256, 0, stream>>>(clfw, clfwT, O, 128, 128);
    }
    {
        dim3 g(1, N / BM);
        gemm_bf16_wmma<<<g, 256, 0, stream>>>(h2_bf, clfwT, clfb, out,
                                              N, 128, O, O, 128);
    }
}
